// DynamicSparseMultiHeadAttention_40381282517698
// MI455X (gfx1250) — matrix operand order repaired and checked by reading, not compiled, not run on hardware
//
#include <hip/hip_runtime.h>
#include <hip/hip_bf16.h>

typedef __attribute__((ext_vector_type(16))) __bf16 v16bf;
typedef __attribute__((ext_vector_type(2)))  __bf16 v2bf;
typedef __attribute__((ext_vector_type(8)))  float  v8f;

#define N_HEADS 8
#define DMODEL  512
#define DK      64
#define WIN     2
#define LQ      2048
#define BATCH   4
#define LDSTR   2052   // fp32 score row stride (2048 + 4 pad)
#define LDSTRB  2056   // bf16 attn row stride in shorts (2048 + 8 pad) -> 1028 dwords
#define NEG_INF (-__builtin_inff())

__device__ __forceinline__ unsigned short f2bf(float f) {
    unsigned int u = __float_as_uint(f);
    u += 0x7FFFu + ((u >> 16) & 1u);           // round to nearest even
    return (unsigned short)(u >> 16);
}

__device__ __forceinline__ unsigned int pack2(float a, float b) {
#if __has_builtin(__builtin_amdgcn_cvt_pk_bf16_f32)
    union { v2bf v; unsigned int u; } c;
    c.v = __builtin_amdgcn_cvt_pk_bf16_f32(a, b);
    return c.u;
#else
    return (unsigned int)f2bf(a) | ((unsigned int)f2bf(b) << 16);
#endif
}

union FragU { v16bf v; unsigned int u[8]; };

__device__ __forceinline__ v8f wmma_bf16(v16bf a, v16bf b, v8f c) {
    return __builtin_amdgcn_wmma_f32_16x16x32_bf16(false, a, false, b, (short)0, c, false, false);
}

// A fragment (16x32 MxK) from bf16 row-major [row*ld + k]; pairs along K contiguous.
// lanes 0-15: M=0..15, VGPR0-3 -> K=0..7, VGPR4-7 -> K=16..23
// lanes 16-31: same M,  VGPR0-3 -> K=8..15, VGPR4-7 -> K=24..31
__device__ __forceinline__ v16bf loadA_bf16(const unsigned short* src, int ld, int mbase, int kstep) {
    int lane = threadIdx.x & 31;
    int half = (lane < 16) ? 0 : 8;
    const unsigned short* p = src + (size_t)(mbase + (lane & 15)) * ld + kstep;
    FragU f;
#pragma unroll
    for (int v = 0; v < 4; ++v) f.u[v]     = *(const unsigned int*)(p + half + 2 * v);
#pragma unroll
    for (int v = 0; v < 4; ++v) f.u[4 + v] = *(const unsigned int*)(p + 16 + half + 2 * v);
    return f.v;
}

// B fragment (32x16 KxN), B[k][n] = src[n*ld + k]; two 16B runs per lane.
// lanes 0-15: N=0..15, K=0..7,16..23 ; lanes 16-31: same N, K=8..15,24..31
__device__ __forceinline__ v16bf loadB_bf16(const unsigned short* src, int ld, int nbase, int kstep) {
    int lane = threadIdx.x & 31;
    int koff = (lane < 16) ? 0 : 8;
    const unsigned int* p =
        (const unsigned int*)(src + (size_t)(nbase + (lane & 15)) * ld + kstep + koff);
    FragU f;
#pragma unroll
    for (int i = 0; i < 8; ++i) f.u[i] = p[i + (i >= 4 ? 4 : 0)];
    return f.v;
}

// ---------------------------------------------------------------------------
// Kernel 0: one-pass fp32 -> bf16 conversion (activations + weights).
// ---------------------------------------------------------------------------
__global__ __launch_bounds__(256) void k_cvt(const float* __restrict__ src,
                                             unsigned short* __restrict__ dst, int n) {
    int i = (blockIdx.x * 256 + threadIdx.x) * 4;
    if (i + 3 < n) {
        float4 v = *(const float4*)(src + i);
        unsigned int* d = (unsigned int*)(dst + i);
        d[0] = pack2(v.x, v.y);
        d[1] = pack2(v.z, v.w);
    }
}

// ---------------------------------------------------------------------------
// Kernel 1: QKV projection (+ bias) -> bf16 tensors; gate mask for Q.
// grid = (512 m-tiles, 3 matrices), block = 256 (8 waves; wave == head)
// ---------------------------------------------------------------------------
__global__ __launch_bounds__(256) void k_proj(
    const unsigned short* __restrict__ xqb, const unsigned short* __restrict__ xkb,
    const unsigned short* __restrict__ xvb,
    const unsigned short* __restrict__ wqb, const unsigned short* __restrict__ wkb,
    const unsigned short* __restrict__ wvb,
    const float* __restrict__ bq, const float* __restrict__ bk, const float* __restrict__ bv,
    const float* __restrict__ wg, const float* __restrict__ bg,
    unsigned short* __restrict__ qbf, unsigned short* __restrict__ kbf,
    unsigned short* __restrict__ vtbf, int* __restrict__ gmask)
{
    const int mtile = blockIdx.x;
    const int mat   = blockIdx.y;            // 0=q 1=k 2=v
    const int wave  = threadIdx.x >> 5;      // == head
    const int lane  = threadIdx.x & 31;
    const int mbase = mtile * 16;
    const int nbase = wave * 64;

    const unsigned short* x    = (mat == 0) ? xqb : ((mat == 1) ? xkb : xvb);
    const unsigned short* W    = (mat == 0) ? wqb : ((mat == 1) ? wkb : wvb);
    const float*          bias = (mat == 0) ? bq  : ((mat == 1) ? bk  : bv);

    v8f acc[4];
#pragma unroll
    for (int t = 0; t < 4; ++t) acc[t] = (v8f){0.f,0.f,0.f,0.f,0.f,0.f,0.f,0.f};

    for (int ks = 0; ks < DMODEL; ks += 32) {
        if (ks + 32 < DMODEL)
            __builtin_prefetch(W + (size_t)(nbase + (lane & 15)) * DMODEL + ks + 32, 0, 1);
        v16bf a = loadA_bf16(x, DMODEL, mbase, ks);
#pragma unroll
        for (int t = 0; t < 4; ++t) {
            v16bf bfr = loadB_bf16(W, DMODEL, nbase + 16 * t, ks);
            acc[t] = wmma_bf16(a, bfr, acc[t]);
        }
    }

    const int half = (lane < 16) ? 0 : 8;
    const int h = wave;
#pragma unroll
    for (int t = 0; t < 4; ++t) {
        float bv_ = bias[nbase + 16 * t + (lane & 15)];
#pragma unroll
        for (int r = 0; r < 8; ++r) acc[t][r] += bv_;
    }

#pragma unroll
    for (int t = 0; t < 4; ++t) {
        int d = 16 * t + (lane & 15);
#pragma unroll
        for (int r = 0; r < 8; ++r) {
            int m = mbase + r + half;
            int b = m >> 11, lp = m & 2047;
            int hb = h * BATCH + b;
            unsigned short bfv = f2bf(acc[t][r]);
            if (mat == 0)      qbf [((size_t)hb * LQ + lp) * DK + d] = bfv;
            else if (mat == 1) kbf [((size_t)hb * LQ + lp) * DK + d] = bfv;
            else               vtbf[((size_t)hb * DK + d) * LQ + lp] = bfv;
        }
    }

    if (mat == 0) {   // gate: sigmoid(qh . w_g + b_g) > 0.5  <=>  logit > 0
        float wgv[4];
#pragma unroll
        for (int t = 0; t < 4; ++t) wgv[t] = wg[16 * t + (lane & 15)];
        float bgv = bg[0];
#pragma unroll
        for (int r = 0; r < 8; ++r) {
            float g = acc[0][r] * wgv[0] + acc[1][r] * wgv[1] +
                      acc[2][r] * wgv[2] + acc[3][r] * wgv[3];
#pragma unroll
            for (int off = 8; off >= 1; off >>= 1) g += __shfl_xor(g, off, 16);
            if ((lane & 15) == 0) {
                int m = mbase + r + half;
                int b = m >> 11, lp = m & 2047;
                gmask[(h * BATCH + b) * LQ + lp] = (g + bgv > 0.0f) ? 1 : 0;
            }
        }
    }
}

// ---------------------------------------------------------------------------
// Kernel 2: fused attention per (hb, 16-row q-tile):
// scores -> LDS, mask, softmax (normalized attn packed to bf16 LDS + streamed
// to global fp32), attn@V with cross-wave reduction.
// grid = (128 q-tiles, 32 hb), block = 256
// ---------------------------------------------------------------------------
__global__ __launch_bounds__(256) void k_attn(
    const unsigned short* __restrict__ qbf, const unsigned short* __restrict__ kbf,
    const unsigned short* __restrict__ vtbf, const int* __restrict__ gmask,
    float* __restrict__ attn_out, unsigned short* __restrict__ o2bf)
{
    __shared__ float        s_sc[16 * LDSTR];        // raw scores / exp values
    __shared__ unsigned int s_bf[16 * (LDSTRB / 2)]; // normalized attn, packed bf16
    __shared__ float        s_red[4 * 256];          // cross-wave partials
    __shared__ int          s_gate[16];

    const int qtile = blockIdx.x;
    const int hb    = blockIdx.y;
    const int wave  = threadIdx.x >> 5;
    const int lane  = threadIdx.x & 31;
    const int qbase = qtile * 16;
    const int half  = (lane < 16) ? 0 : 8;

    if (threadIdx.x < 16) s_gate[threadIdx.x] = gmask[hb * LQ + qbase + threadIdx.x];
    __syncthreads();

    const unsigned short* qrow = qbf + (size_t)hb * LQ * DK;
    const unsigned short* krow = kbf + (size_t)hb * LQ * DK;

    v16bf qf0 = loadA_bf16(qrow, DK, qbase, 0);
    v16bf qf1 = loadA_bf16(qrow, DK, qbase, 32);

    // ---- phase A: masked scaled scores into LDS -----------------------------
    for (int ct = 0; ct < 16; ++ct) {
        int nc = wave * 256 + ct * 16;
        if (ct + 1 < 16)
            __builtin_prefetch(krow + (size_t)(nc + 16 + (lane & 15)) * DK, 0, 1);
        v16bf b0 = loadB_bf16(krow, DK, nc, 0);
        v16bf b1 = loadB_bf16(krow, DK, nc, 32);
        v8f acc = (v8f){0.f,0.f,0.f,0.f,0.f,0.f,0.f,0.f};
        acc = wmma_bf16(qf0, b0, acc);
        acc = wmma_bf16(qf1, b1, acc);
        int kcol = nc + (lane & 15);
#pragma unroll
        for (int r = 0; r < 8; ++r) {
            int ml = r + half;
            float s = acc[r] * 0.125f;                    // 1/sqrt(64)
            int dd = (qbase + ml) - kcol; if (dd < 0) dd = -dd;
            bool allowed = (s_gate[ml] != 0) || (dd <= WIN);
            s_sc[ml * LDSTR + kcol] = allowed ? s : NEG_INF;
        }
    }
    __syncthreads();

    // ---- phase B: row softmax; bf16-pack normalized attn to LDS + global ----
    {
        int row = threadIdx.x >> 4;
        int seg = threadIdx.x & 15;
        float* rp = s_sc + row * LDSTR + seg * 128;
        float mx = NEG_INF;
        for (int i = 0; i < 128; i += 4) {
            float4 v = *(float4*)(rp + i);
            mx = fmaxf(mx, fmaxf(fmaxf(v.x, v.y), fmaxf(v.z, v.w)));
        }
#pragma unroll
        for (int off = 8; off >= 1; off >>= 1) mx = fmaxf(mx, __shfl_xor(mx, off, 16));
        float sum = 0.f;
        for (int i = 0; i < 128; i += 4) {
            float4 v = *(float4*)(rp + i);
            v.x = __expf(v.x - mx); v.y = __expf(v.y - mx);
            v.z = __expf(v.z - mx); v.w = __expf(v.w - mx);
            *(float4*)(rp + i) = v;
            sum += v.x + v.y + v.z + v.w;
        }
#pragma unroll
        for (int off = 8; off >= 1; off >>= 1) sum += __shfl_xor(sum, off, 16);
        float inv = 1.0f / sum;
        float* gp = attn_out + ((size_t)hb * LQ + (qbase + row)) * LQ + seg * 128;
        unsigned int* bp = s_bf + row * (LDSTRB / 2) + seg * 64;
        for (int i = 0; i < 128; i += 4) {
            float4 v = *(float4*)(rp + i);
            v.x *= inv; v.y *= inv; v.z *= inv; v.w *= inv;
            *(float4*)(gp + i) = v;
            bp[i / 2]     = pack2(v.x, v.y);
            bp[i / 2 + 1] = pack2(v.z, v.w);
        }
    }
    __syncthreads();

    // ---- phase C: out2 = attn @ V (A from bf16 LDS, B from transposed V) ----
    const int dtile = wave & 3;
    const int khalf = wave >> 2;
    const int dbase = dtile * 16;
    const unsigned short* vrow = vtbf + (size_t)hb * DK * LQ;
    const unsigned short* abf  = (const unsigned short*)s_bf;

    v8f acc = (v8f){0.f,0.f,0.f,0.f,0.f,0.f,0.f,0.f};
    for (int ks = khalf * 1024; ks < khalf * 1024 + 1024; ks += 32) {
        v16bf a = loadA_bf16(abf, LDSTRB, 0, ks);          // plain ds_load_b32
        v16bf b = loadB_bf16(vrow, LQ, dbase, ks);
        acc = wmma_bf16(a, b, acc);
    }
    if (khalf == 1) {
#pragma unroll
        for (int r = 0; r < 8; ++r)
            s_red[dtile * 256 + (r + half) * 16 + (lane & 15)] = acc[r];
    }
    __syncthreads();
    if (khalf == 0) {
        int h = hb >> 2, b = hb & 3;
#pragma unroll
        for (int r = 0; r < 8; ++r) {
            int ml = r + half;
            float v = acc[r] + s_red[dtile * 256 + ml * 16 + (lane & 15)];
            size_t grow = (size_t)b * LQ + (qbase + ml);
            int col = h * DK + dbase + (lane & 15);
            o2bf[grow * DMODEL + col] = f2bf(v);
        }
    }
}

// ---------------------------------------------------------------------------
// Kernel 3: fc projection + bias + residual + LayerNorm.
// grid = 512 m-tiles, block = 256 (8 waves x 64 columns)
// ---------------------------------------------------------------------------
__global__ __launch_bounds__(256) void k_fc_ln(
    const unsigned short* __restrict__ o2bf, const unsigned short* __restrict__ wfcb,
    const float* __restrict__ bfc, const float* __restrict__ resid,
    const float* __restrict__ lng, const float* __restrict__ lnb,
    float* __restrict__ out)
{
    __shared__ float s_sum[16];
    __shared__ float s_sq[16];
    if (threadIdx.x < 16) { s_sum[threadIdx.x] = 0.f; s_sq[threadIdx.x] = 0.f; }

    const int mtile = blockIdx.x;
    const int wave  = threadIdx.x >> 5;
    const int lane  = threadIdx.x & 31;
    const int mbase = mtile * 16;
    const int nbase = wave * 64;
    const int half  = (lane < 16) ? 0 : 8;

    v8f acc[4];
#pragma unroll
    for (int t = 0; t < 4; ++t) acc[t] = (v8f){0.f,0.f,0.f,0.f,0.f,0.f,0.f,0.f};

    for (int ks = 0; ks < DMODEL; ks += 32) {
        v16bf a = loadA_bf16(o2bf, DMODEL, mbase, ks);
#pragma unroll
        for (int t = 0; t < 4; ++t) {
            v16bf bfr = loadB_bf16(wfcb, DMODEL, nbase + 16 * t, ks);
            acc[t] = wmma_bf16(a, bfr, acc[t]);
        }
    }

    float xv[4][8];
#pragma unroll
    for (int t = 0; t < 4; ++t) {
        int n = nbase + 16 * t + (lane & 15);
        float bb = bfc[n];
#pragma unroll
        for (int r = 0; r < 8; ++r) {
            int m = mbase + r + half;
            xv[t][r] = acc[t][r] + bb + resid[(size_t)m * DMODEL + n];
        }
    }
    __syncthreads();   // zeroed stats visible before atomics

#pragma unroll
    for (int r = 0; r < 8; ++r) {
        float p  = xv[0][r] + xv[1][r] + xv[2][r] + xv[3][r];
        float q2 = xv[0][r] * xv[0][r] + xv[1][r] * xv[1][r] +
                   xv[2][r] * xv[2][r] + xv[3][r] * xv[3][r];
#pragma unroll
        for (int off = 8; off >= 1; off >>= 1) {
            p  += __shfl_xor(p, off, 16);
            q2 += __shfl_xor(q2, off, 16);
        }
        if ((lane & 15) == 0) {
            atomicAdd(&s_sum[r + half], p);    // ds_add_f32
            atomicAdd(&s_sq [r + half], q2);
        }
    }
    __syncthreads();

#pragma unroll
    for (int t = 0; t < 4; ++t) {
        int n = nbase + 16 * t + (lane & 15);
        float g = lng[n], bta = lnb[n];
#pragma unroll
        for (int r = 0; r < 8; ++r) {
            int ml = r + half;
            float mean = s_sum[ml] * (1.0f / 512.0f);
            float var  = s_sq [ml] * (1.0f / 512.0f) - mean * mean;
            float inv  = rsqrtf(var + 1e-5f);
            int m = mbase + ml;
            out[(size_t)m * DMODEL + n] = (xv[t][r] - mean) * inv * g + bta;
        }
    }
}

// ---------------------------------------------------------------------------
extern "C" void kernel_launch(void* const* d_in, const int* in_sizes, int n_in,
                              void* d_out, int out_size, void* d_ws, size_t ws_size,
                              hipStream_t stream) {
    const float* q    = (const float*)d_in[0];
    const float* k    = (const float*)d_in[1];
    const float* v    = (const float*)d_in[2];
    const float* w_q  = (const float*)d_in[3];
    const float* b_q  = (const float*)d_in[4];
    const float* w_k  = (const float*)d_in[5];
    const float* b_k  = (const float*)d_in[6];
    const float* w_v  = (const float*)d_in[7];
    const float* b_v  = (const float*)d_in[8];
    const float* w_g  = (const float*)d_in[9];
    const float* b_g  = (const float*)d_in[10];
    const float* w_fc = (const float*)d_in[11];
    const float* b_fc = (const float*)d_in[12];
    const float* ln_g = (const float*)d_in[13];
    const float* ln_b = (const float*)d_in[14];

    float* out  = (float*)d_out;
    float* attn = out + (size_t)BATCH * LQ * DMODEL;   // out first, then attn

    char* ws = (char*)d_ws;
    size_t off = 0;
    const size_t act  = (size_t)BATCH * LQ * DMODEL * sizeof(unsigned short); // 8 MB
    const size_t wsz  = (size_t)DMODEL * DMODEL * sizeof(unsigned short);     // 0.5 MB
    unsigned short* qbf   = (unsigned short*)(ws + off); off += act;
    unsigned short* kbf   = (unsigned short*)(ws + off); off += act;
    unsigned short* vtbf  = (unsigned short*)(ws + off); off += act;
    unsigned short* o2bf  = (unsigned short*)(ws + off); off += act;
    unsigned short* xqb   = (unsigned short*)(ws + off); off += act;
    unsigned short* xkb   = (unsigned short*)(ws + off); off += act;
    unsigned short* xvb   = (unsigned short*)(ws + off); off += act;
    unsigned short* wqb   = (unsigned short*)(ws + off); off += wsz;
    unsigned short* wkb   = (unsigned short*)(ws + off); off += wsz;
    unsigned short* wvb   = (unsigned short*)(ws + off); off += wsz;
    unsigned short* wfcb  = (unsigned short*)(ws + off); off += wsz;
    int* gmask = (int*)(ws + off);

    const int nAct = BATCH * LQ * DMODEL;    // 4,194,304
    const int nW   = DMODEL * DMODEL;        // 262,144
    k_cvt<<<nAct / (256 * 4), 256, 0, stream>>>(q, xqb, nAct);
    k_cvt<<<nAct / (256 * 4), 256, 0, stream>>>(k, xkb, nAct);
    k_cvt<<<nAct / (256 * 4), 256, 0, stream>>>(v, xvb, nAct);
    k_cvt<<<nW / (256 * 4), 256, 0, stream>>>(w_q, wqb, nW);
    k_cvt<<<nW / (256 * 4), 256, 0, stream>>>(w_k, wkb, nW);
    k_cvt<<<nW / (256 * 4), 256, 0, stream>>>(w_v, wvb, nW);
    k_cvt<<<nW / (256 * 4), 256, 0, stream>>>(w_fc, wfcb, nW);

    k_proj<<<dim3(512, 3), 256, 0, stream>>>(xqb, xkb, xvb, wqb, wkb, wvb,
                                             b_q, b_k, b_v, w_g, b_g,
                                             qbf, kbf, vtbf, gmask);
    k_attn<<<dim3(128, 32), 256, 0, stream>>>(qbf, kbf, vtbf, gmask, attn, o2bf);
    k_fc_ln<<<dim3(512), 256, 0, stream>>>(o2bf, w_fc ? wfcb : wfcb, b_fc, q, ln_g, ln_b, out);
}
